// ConditionalTagDecoder_18537078850186
// MI455X (gfx1250) — matrix operand order repaired and checked by reading, not compiled, not run on hardware
//
#include <hip/hip_runtime.h>
#include <hip/hip_bf16.h>
#include <cstdint>
#include <cstddef>

// Problem dims
#define B_  2048
#define L_  512
#define E_  64
#define H_  128
#define D_  256
#define V_  1000

// Attention chunking
#define LC        128                 // rows of enc per LDS chunk
#define NCHUNK    (L_ / LC)           // 4
#define ROWP      (D_ + 4)            // padded LDS row stride (dwords): bank spread for row gathers
#define CHUNK_DW  (LC * ROWP)         // 33280 dwords per buffer
#define SA_DW     (8 * 32 * 8)        // bf16 A-panel staging: 8 kt x 32 lanes x 8 dwords = 2048

typedef __bf16 bf16_t;
typedef bf16_t       v16bf  __attribute__((ext_vector_type(16)));
typedef bf16_t       bf16x2 __attribute__((ext_vector_type(2)));
typedef float        v8f    __attribute__((ext_vector_type(8)));
typedef unsigned int u32x4  __attribute__((ext_vector_type(4)));
typedef int          i32x4  __attribute__((ext_vector_type(4)));
typedef int          i32x8  __attribute__((ext_vector_type(8)));

// ---- fast math on hardware transcendentals (no branchy libm paths) ----
#define LOG2E  1.4426950408889634f

__device__ __forceinline__ float exp_fast(float x) {          // e^x, x <= ~0 here
  return __builtin_amdgcn_exp2f(x * LOG2E);
}
__device__ __forceinline__ float tanh_fast(float x) {
  x = fminf(fmaxf(x, -15.f), 15.f);
  const float t = __builtin_amdgcn_exp2f(x * (2.f * LOG2E));
  return (t - 1.f) * __builtin_amdgcn_rcpf(t + 1.f);
}
__device__ __forceinline__ float sigmoid_fast(float x) {
  return __builtin_amdgcn_rcpf(1.f + __builtin_amdgcn_exp2f(-x * LOG2E));
}

// ---- native f32 -> bf16 conversion (lowers to v_cvt_*bf16* on gfx1250) ----
__device__ __forceinline__ bf16_t f2bf(float x) { return (bf16_t)x; }
__device__ __forceinline__ unsigned pack2bf(float a, float b) {
  bf16x2 r = {(bf16_t)a, (bf16_t)b};
  return __builtin_bit_cast(unsigned, r);
}

// ---------------------------------------------------------------------------
// Tensor Data Mover: DMA a (LC x D_) fp32 tile into LDS, padding each 256-dword
// row by 4 dwords (ROWP=260) via the TDM pad feature.
// ---------------------------------------------------------------------------
__device__ __forceinline__ void tdm_load_chunk(uint64_t gaddr, unsigned lds_byte_off) {
  u32x4 g0;
  g0[0] = 1u;                                         // count=1 valid descriptor
  g0[1] = lds_byte_off;                               // lds_addr
  g0[2] = (unsigned)(gaddr & 0xFFFFFFFFull);          // global_addr[31:0]
  g0[3] = (unsigned)((gaddr >> 32) & 0x1FFFFFFull)    // global_addr[56:32]
        | (2u << 30);                                 // type = 2 ("image")
  i32x8 g1;
  g1[0] = (int)((2u << 16)        // data_size = 4 bytes
              | (1u << 20)        // pad_enable
              | (7u << 22)        // pad_interval code 7 -> every 256 dwords
              | (3u << 25));      // pad_amount  code 3 -> 4 dwords
  g1[1] = (int)((unsigned)D_ << 16);   // tensor_dim0[15:0] = 256
  g1[2] = (int)((unsigned)LC << 16);   // tensor_dim1[15:0] = 128 (tile in-bounds)
  g1[3] = (int)((unsigned)D_ << 16);   // tile_dim0 = 256
  g1[4] = (int)LC;                     // tile_dim1 = 128, tile_dim2 = 0
  g1[5] = (int)D_;                     // tensor_dim0_stride = 256 (contiguous rows)
  g1[6] = 0;
  g1[7] = 0;
  i32x4 z4 = {0, 0, 0, 0};
#if defined(__clang_major__) && (__clang_major__ >= 23)
  i32x8 z8 = {0, 0, 0, 0, 0, 0, 0, 0};
  __builtin_amdgcn_tensor_load_to_lds(g0, g1, z4, z4, z8, 0);
#else
  __builtin_amdgcn_tensor_load_to_lds(g0, g1, z4, z4, 0);
#endif
}

// ---------------------------------------------------------------------------
// Kernel 1: qproj[b,h] = (hidden[1,b,:] @ Wq^T)[h]    (B x H, tiny)
// ---------------------------------------------------------------------------
__global__ __launch_bounds__(H_) void qproj_kernel(const float* __restrict__ hidden,
                                                   const float* __restrict__ Wq,
                                                   float* __restrict__ qproj) {
  const int b = blockIdx.x, t = threadIdx.x;
  __shared__ float q[H_];
  q[t] = hidden[(size_t)B_ * H_ + (size_t)b * H_ + t];
  __syncthreads();
  const float* w = Wq + (size_t)t * H_;
  float acc = 0.f;
#pragma unroll 4
  for (int k = 0; k < H_; ++k) acc += q[k] * w[k];
  qproj[(size_t)b * H_ + t] = acc;
}

// ---------------------------------------------------------------------------
// Kernel 2: fused Bahdanau attention, one workgroup (8 waves) per batch row.
// TDM double-buffered fp32 chunks -> per-M-panel cooperative bf16 staging in
// WMMA fragment order -> 8 back-to-back v_wmma_f32_16x16x32_bf16 per panel ->
// tanh/dot(v) scores -> online softmax -> context. One HBM pass over enc.
// ---------------------------------------------------------------------------
__global__ __launch_bounds__(256) void attn_kernel(const float* __restrict__ enc,
                                                   const float* __restrict__ qproj,
                                                   const float* __restrict__ vvec,
                                                   const float* __restrict__ Wk,
                                                   float* __restrict__ attn_out,
                                                   float* __restrict__ ctx_out) {
  extern __shared__ float sm[];
  float*    encs   = sm;                                 // 2 * CHUNK_DW (double buffer)
  unsigned* sA     = (unsigned*)(sm + 2 * CHUNK_DW);     // SA_DW: bf16 A-panel staging
  float*    scores = sm + 2 * CHUNK_DW + SA_DW;          // L_ raw scores
  float*    pbuf   = scores + L_;                        // LC
  float*    red    = pbuf + LC;                          // LC
  float*    qv     = red + LC;                           // H_
  float*    vv     = qv + H_;                            // H_
  float*    scal   = vv + H_;                            // m_run, S_run, factor, m_new

  const int b    = blockIdx.x;
  const int t    = threadIdx.x;
  const int wave = t >> 5;
  const int lane = t & 31;
  const int lr   = lane & 15;
  const int lh   = lane >> 4;

  if (t < H_) { qv[t] = qproj[(size_t)b * H_ + t]; vv[t] = vvec[t]; }
  for (int i = t; i < L_; i += 256) scores[i] = 0.f;
  if (t == 0) { scal[0] = -__builtin_inff(); scal[1] = 0.f; }

  // Preload this wave's Wk^T B-fragments (n-tile = wave, 8 k-tiles) as bf16.
  // B frag in this code's contiguous k order (A matches): lane holds col N=lane%16; elem e -> K = 16*(lane/16)+e.
  v16bf bfrag[8];
  {
    const int n = wave * 16 + lr;
    const float* wkrow = Wk + (size_t)n * D_;   // Wk is (H,D); B[k][n] = Wk[n][k]
#pragma unroll
    for (int kt = 0; kt < 8; ++kt) {
      v16bf f;
#pragma unroll
      for (int e = 0; e < 16; ++e) f[e] = f2bf(wkrow[kt * 32 + lh * 16 + e]);
      bfrag[kt] = f;
    }
  }

  const uint64_t gbase = (uint64_t)(uintptr_t)enc + (uint64_t)b * ((uint64_t)L_ * D_ * 4);
  if (wave == 0) tdm_load_chunk(gbase, 0u);   // prologue: chunk 0 -> buffer 0

  float cacc = 0.f;   // context accumulator; thread t owns column d = t

  for (int c = 0; c < NCHUNK; ++c) {
    if (wave == 0) {
      if (c + 1 < NCHUNK) {
        tdm_load_chunk(gbase + (uint64_t)(c + 1) * LC * D_ * 4,
                       (unsigned)(((c + 1) & 1) * CHUNK_DW * 4));
        __builtin_amdgcn_s_wait_tensorcnt(1);   // chunk c landed, c+1 in flight
      } else {
        __builtin_amdgcn_s_wait_tensorcnt(0);
      }
    }
    __syncthreads();
    const float* ec = encs + (size_t)(c & 1) * CHUNK_DW;

    const float qvn = qv[wave * 16 + lr];
    const float vn  = vv[wave * 16 + lr];

    for (int mt = 0; mt < 8; ++mt) {
      // ---- cooperative bf16 staging of A panel (16 rows x 256 cols), fragment order.
      // thread t handles (kt = t/32, lane = t%32): 8 packed dwords.
      {
        const int ktc = wave;            // == t >> 5
        const int m   = mt * 16 + lr;
        const float* row = ec + (size_t)m * ROWP;
        unsigned* dst = sA + (ktc * 32 + lane) * 8;
#pragma unroll
        for (int j = 0; j < 8; ++j) {
          // dword j holds elems e=2j,2j+1: K = kt*32 + 16*lh + e
          const int K = ktc * 32 + (lh << 4) + 2 * j;
          dst[j] = pack2bf(row[K], row[K + 1]);
        }
      }
      __syncthreads();

      // ---- preload all 8 A fragments, then 8 back-to-back WMMAs ----
      const unsigned* afrag = sA + lane * 8;
      v16bf afr[8];
#pragma unroll
      for (int kt = 0; kt < 8; ++kt) afr[kt] = *(const v16bf*)(afrag + kt * 256);
      v8f acc = {};
#pragma unroll
      for (int kt = 0; kt < 8; ++kt)
        acc = __builtin_amdgcn_wmma_f32_16x16x32_bf16(false, afr[kt], false, bfrag[kt],
                                                      (short)0, acc, false, false);

      // score[m] += sum_n tanh(energy[m,n]) * v[n]; butterfly over 16 column-lanes,
      // step-major so 8 independent bpermutes are in flight per wait.
      float p[8];
#pragma unroll
      for (int j = 0; j < 8; ++j) p[j] = tanh_fast(acc[j] + qvn) * vn;
#pragma unroll
      for (int ofs = 1; ofs < 16; ofs <<= 1) {
#pragma unroll
        for (int j = 0; j < 8; ++j) p[j] += __shfl_xor(p[j], ofs, 16);
      }
      if (lr == 0) {
#pragma unroll
        for (int j = 0; j < 8; ++j)
          atomicAdd(&scores[c * LC + mt * 16 + j + (lh << 3)], p[j]);
      }
      __syncthreads();   // sA consumed; safe to overwrite next panel
    }

    // ---- online softmax update ----
    if (t < LC) red[t] = scores[c * LC + t];
    __syncthreads();
    for (int s = LC / 2; s > 0; s >>= 1) {
      if (t < s) red[t] = fmaxf(red[t], red[t + s]);
      __syncthreads();
    }
    if (t == 0) {
      const float mo = scal[0];
      const float mn = fmaxf(mo, red[0]);
      scal[3] = mn;
      scal[2] = exp_fast(mo - mn);   // == 0 on first chunk (mo = -inf)
      scal[0] = mn;
    }
    __syncthreads();
    const float mn = scal[3], factor = scal[2];
    if (t < LC) { pbuf[t] = exp_fast(scores[c * LC + t] - mn); red[t] = pbuf[t]; }
    __syncthreads();
    for (int s = LC / 2; s > 0; s >>= 1) {
      if (t < s) red[t] += red[t + s];
      __syncthreads();
    }
    if (t == 0) scal[1] = scal[1] * factor + red[0];

    // ---- context accumulation: thread t owns column d = t (no bank conflicts) ----
    float ca = cacc * factor;
#pragma unroll 4
    for (int l = 0; l < LC; ++l) ca += pbuf[l] * ec[(size_t)l * ROWP + t];
    cacc = ca;
    __syncthreads();   // buffer (c&1) free for reuse; scal updates visible
  }

  const float mfin = scal[0];
  const float inv  = __builtin_amdgcn_rcpf(scal[1]);
  ctx_out[(size_t)b * D_ + t] = cacc * inv;
  for (int l = t; l < L_; l += 256)
    attn_out[(size_t)b * L_ + l] = exp_fast(scores[l] - mfin) * inv;
}

// ---------------------------------------------------------------------------
// Kernel 3: 2-layer GRU cell, one block per batch row (sequential gate math).
// ---------------------------------------------------------------------------
__global__ __launch_bounds__(H_) void gru_kernel(
    const float* __restrict__ embed, const float* __restrict__ hidden,
    const float* __restrict__ ctx,
    const float* __restrict__ Wih0, const float* __restrict__ Whh0,
    const float* __restrict__ bih0, const float* __restrict__ bhh0,
    const float* __restrict__ Wih1, const float* __restrict__ Whh1,
    const float* __restrict__ bih1, const float* __restrict__ bhh1,
    float* __restrict__ newhid) {
  const int b = blockIdx.x, t = threadIdx.x;
  __shared__ float x0[E_ + D_];
  __shared__ float h0o[H_], h1o[H_], h0n[H_];
  if (t < E_) x0[t] = embed[(size_t)b * E_ + t];
  for (int i = t; i < D_; i += H_) x0[E_ + i] = ctx[(size_t)b * D_ + i];
  h0o[t] = hidden[(size_t)b * H_ + t];
  h1o[t] = hidden[(size_t)B_ * H_ + (size_t)b * H_ + t];
  __syncthreads();
  {  // layer 0 (input = [embed, context], K = 320)
    const int KI = E_ + D_;
    float gr = bih0[t], gz = bih0[H_ + t], gn = bih0[2 * H_ + t];
    const float* wr = Wih0 + (size_t)t * KI;
    const float* wz = Wih0 + (size_t)(H_ + t) * KI;
    const float* wn = Wih0 + (size_t)(2 * H_ + t) * KI;
    for (int k = 0; k < KI; ++k) { const float x = x0[k]; gr += x * wr[k]; gz += x * wz[k]; gn += x * wn[k]; }
    float hr = bhh0[t], hz = bhh0[H_ + t], hn = bhh0[2 * H_ + t];
    const float* ur = Whh0 + (size_t)t * H_;
    const float* uz = Whh0 + (size_t)(H_ + t) * H_;
    const float* un = Whh0 + (size_t)(2 * H_ + t) * H_;
    for (int k = 0; k < H_; ++k) { const float h = h0o[k]; hr += h * ur[k]; hz += h * uz[k]; hn += h * un[k]; }
    const float r = sigmoid_fast(gr + hr);
    const float z = sigmoid_fast(gz + hz);
    const float n = tanh_fast(gn + r * hn);
    const float h0 = (1.f - z) * n + z * h0o[t];
    h0n[t] = h0;
    newhid[(size_t)b * H_ + t] = h0;
  }
  __syncthreads();
  {  // layer 1 (input = h0_new, K = 128)
    float gr = bih1[t], gz = bih1[H_ + t], gn = bih1[2 * H_ + t];
    const float* wr = Wih1 + (size_t)t * H_;
    const float* wz = Wih1 + (size_t)(H_ + t) * H_;
    const float* wn = Wih1 + (size_t)(2 * H_ + t) * H_;
    float hr = bhh1[t], hz = bhh1[H_ + t], hn = bhh1[2 * H_ + t];
    const float* ur = Whh1 + (size_t)t * H_;
    const float* uz = Whh1 + (size_t)(H_ + t) * H_;
    const float* un = Whh1 + (size_t)(2 * H_ + t) * H_;
    for (int k = 0; k < H_; ++k) {
      const float x = h0n[k], h = h1o[k];
      gr += x * wr[k]; gz += x * wz[k]; gn += x * wn[k];
      hr += h * ur[k]; hz += h * uz[k]; hn += h * un[k];
    }
    const float r = sigmoid_fast(gr + hr);
    const float z = sigmoid_fast(gz + hz);
    const float n = tanh_fast(gn + r * hn);
    newhid[(size_t)B_ * H_ + (size_t)b * H_ + t] = (1.f - z) * n + z * h1o[t];
  }
}

// ---------------------------------------------------------------------------
// Kernel 4: logits = h1 @ W_out^T + b_out via bf16 WMMA (M = batch tiles).
// ---------------------------------------------------------------------------
__global__ __launch_bounds__(256) void logits_kernel(const float* __restrict__ h1,
                                                     const float* __restrict__ Wout,
                                                     const float* __restrict__ bout,
                                                     float* __restrict__ out) {
  const int t = threadIdx.x, wave = t >> 5, lane = t & 31;
  const int lr = lane & 15, lh = lane >> 4;
  const int m0 = blockIdx.x * 16;
  const int n0 = (blockIdx.y * 8 + wave) * 16;
  const int n  = n0 + lr;

  v8f acc = {};
  const float* hrow = h1 + (size_t)(m0 + lr) * H_;
  const float* wrow = Wout + (size_t)n * H_;   // W_out is (V,H); B[k][n] = W_out[n][k]
#pragma unroll
  for (int kt = 0; kt < 4; ++kt) {
    v16bf a, bf;
#pragma unroll
    for (int e = 0; e < 16; ++e) {
      const int K = kt * 32 + (lh << 4) + e;
      a[e] = f2bf(hrow[K]);
    }
#pragma unroll
    for (int e = 0; e < 16; ++e) {
      const int K = kt * 32 + (lh << 4) + e;
      bf[e] = (n < V_) ? f2bf(wrow[K]) : f2bf(0.f);
    }
    acc = __builtin_amdgcn_wmma_f32_16x16x32_bf16(false, a, false, bf,
                                                  (short)0, acc, false, false);
  }
  if (n < V_) {
    const float bo = bout[n];
#pragma unroll
    for (int j = 0; j < 8; ++j) {
      const int m = m0 + j + (lh << 3);
      out[(size_t)m * V_ + n] = acc[j] + bo;
    }
  }
}

// ---------------------------------------------------------------------------
extern "C" void kernel_launch(void* const* d_in, const int* in_sizes, int n_in,
                              void* d_out, int out_size, void* d_ws, size_t ws_size,
                              hipStream_t stream) {
  const float* embed  = (const float*)d_in[0];
  const float* hidden = (const float*)d_in[1];
  const float* enc    = (const float*)d_in[2];
  const float* Wq     = (const float*)d_in[3];
  const float* Wk     = (const float*)d_in[4];
  const float* v      = (const float*)d_in[5];
  const float* Wih0   = (const float*)d_in[6];
  const float* Whh0   = (const float*)d_in[7];
  const float* bih0   = (const float*)d_in[8];
  const float* bhh0   = (const float*)d_in[9];
  const float* Wih1   = (const float*)d_in[10];
  const float* Whh1   = (const float*)d_in[11];
  const float* bih1   = (const float*)d_in[12];
  const float* bhh1   = (const float*)d_in[13];
  const float* Wout   = (const float*)d_in[14];
  const float* bout   = (const float*)d_in[15];

  float* out    = (float*)d_out;
  float* logits = out;                                   // (B, V)
  float* newhid = out + (size_t)B_ * V_;                 // (2, B, H)
  float* attnw  = newhid + 2ull * B_ * H_;               // (B, L)

  float* ws    = (float*)d_ws;
  float* qproj = ws;                                     // (B, H)
  float* ctx   = ws + (size_t)B_ * H_;                   // (B, D)

  qproj_kernel<<<B_, H_, 0, stream>>>(hidden, Wq, qproj);

  const size_t smem_bytes =
      (size_t)(2 * CHUNK_DW + SA_DW + L_ + LC + LC + H_ + H_ + 8) * sizeof(float);
  attn_kernel<<<B_, 256, smem_bytes, stream>>>(enc, qproj, v, Wk, attnw, ctx);

  gru_kernel<<<B_, H_, 0, stream>>>(embed, hidden, ctx,
                                    Wih0, Whh0, bih0, bhh0,
                                    Wih1, Whh1, bih1, bhh1, newhid);

  logits_kernel<<<dim3(B_ / 16, 8), 256, 0, stream>>>(newhid + (size_t)B_ * H_,
                                                      Wout, bout, logits);
}
